// HanumanO1Model_74560632259198
// MI455X (gfx1250) — compile-verified, matrix operand order checked by reading, not run on hardware
//
#include <hip/hip_runtime.h>
#include <hip/hip_bf16.h>
#include <math.h>

#define D_    768
#define H_    12
#define DK_   64
#define FF_   3072
#define B_    4
#define S_    512
#define NLAY  12
#define NREAS 3
#define V_    32000
#define RELV  512
#define M_    (B_*S_)

typedef __bf16 bf16_t;
typedef __bf16 v16bf __attribute__((ext_vector_type(16)));
typedef float  v8f   __attribute__((ext_vector_type(8)));

// ---------------------------------------------------------------------------
// Generic batched GEMM: C[M,N] = A[M,K] @ B[K,N] (+bias), fp32 in/out,
// bf16 WMMA (v_wmma_f32_16x16x32_bf16) compute. 256 thr = 8 wave32,
// macro tile 128(M) x 64(N) x 64(K-step); each wave owns a 32x32 patch
// (2x2 fragments, 8 WMMAs per K-step). Register-double-buffered staging:
// next K-tile's global loads issue while current tile's WMMAs run.
// bTrans==0: Bm is [K][ldb] row-major (n contiguous)
// bTrans==1: Bm is [N][ldb] row-major (k contiguous)  (i.e. B = Bm^T)
// Batch z = zo*innerCount + zi ; operand base += zo*Out + zi*In strides.
// Requires: M%128==0, N%64==0, K%64==0, lda/ldb %4==0. All call sites comply.
// ---------------------------------------------------------------------------
__global__ __launch_bounds__(256) void gemm_bf16_wmma(
    const float* __restrict__ A, const float* __restrict__ Bm,
    float* __restrict__ C, const float* __restrict__ bias,
    int K, int lda, int ldb, int ldc,
    long aOut, long aIn, long bOut, long bIn, long cOut, long cIn,
    int innerCount, int bTrans)
{
  __shared__ bf16_t sA[128][66];   // 64 K-cols + pad -> 33 dwords/row (odd)
  __shared__ bf16_t sB[64][66];    // stored N-major: sB[n][k]

  const int tid  = threadIdx.x;
  const int lane = tid & 31;
  const int wave = tid >> 5;
  const int wm   = (wave & 3) * 32;
  const int wn   = (wave >> 2) * 32;
  const long tileN0 = (long)blockIdx.x * 64;
  const long tileM0 = (long)blockIdx.y * 128;

  const int z  = blockIdx.z;
  const int zo = z / innerCount;
  const int zi = z - zo * innerCount;
  A  += (long)zo * aOut + (long)zi * aIn;
  Bm += (long)zo * bOut + (long)zi * bIn;
  C  += (long)zo * cOut + (long)zi * cIn;

  const int lrow = lane & 15;
  const int kcol = (lane >> 4) * 8;   // 0 or 8 (ISA 16-bit A/B lane layout)

  v8f zero = {0.f,0.f,0.f,0.f,0.f,0.f,0.f,0.f};
  v8f acc[2][2] = {{zero, zero}, {zero, zero}};

  float4 ra[8];    // A tile prefetch: 128x64 / 256thr = 8 float4 each
  float4 rb[4];    // B tile prefetch: 64x64 / 256thr = 4 float4 each

  // ---- batched tile loads (all loads issue before any use -> MLP) ----
  auto loadA = [&](int k0) {
#pragma unroll
    for (int i = 0; i < 8; ++i) {
      const int l = (tid + i*256) << 2;       // 0..8191
      const int r = l >> 6, c = l & 63;
      ra[i] = *(const float4*)(A + (tileM0 + r)*(long)lda + (k0 + c));
    }
  };
  auto loadB = [&](int k0) {
    if (bTrans) {
#pragma unroll
      for (int i = 0; i < 4; ++i) {
        const int l = (tid + i*256) << 2;     // 0..4095
        const int n = l >> 6, c = l & 63;
        rb[i] = *(const float4*)(Bm + (tileN0 + n)*(long)ldb + (k0 + c));
      }
    } else {
#pragma unroll
      for (int i = 0; i < 4; ++i) {
        const int l = (tid + i*256) << 2;
        const int kk = l >> 6, n = l & 63;
        rb[i] = *(const float4*)(Bm + (long)(k0 + kk)*ldb + tileN0 + n);
      }
    }
  };
  auto stageLDS = [&]() {
#pragma unroll
    for (int i = 0; i < 8; ++i) {
      const int l = (tid + i*256) << 2;
      const int r = l >> 6, c = l & 63;
      sA[r][c+0] = (bf16_t)ra[i].x; sA[r][c+1] = (bf16_t)ra[i].y;
      sA[r][c+2] = (bf16_t)ra[i].z; sA[r][c+3] = (bf16_t)ra[i].w;
    }
    if (bTrans) {
#pragma unroll
      for (int i = 0; i < 4; ++i) {
        const int l = (tid + i*256) << 2;
        const int n = l >> 6, c = l & 63;
        sB[n][c+0] = (bf16_t)rb[i].x; sB[n][c+1] = (bf16_t)rb[i].y;
        sB[n][c+2] = (bf16_t)rb[i].z; sB[n][c+3] = (bf16_t)rb[i].w;
      }
    } else {
#pragma unroll
      for (int i = 0; i < 4; ++i) {
        const int l = (tid + i*256) << 2;
        const int kk = l >> 6, n = l & 63;
        sB[n+0][kk] = (bf16_t)rb[i].x; sB[n+1][kk] = (bf16_t)rb[i].y;
        sB[n+2][kk] = (bf16_t)rb[i].z; sB[n+3][kk] = (bf16_t)rb[i].w;
      }
    }
  };

  loadA(0); loadB(0);

  for (int k0 = 0; k0 < K; k0 += 64) {
    stageLDS();
    __syncthreads();

    // prefetch next K-tile while this tile computes
    if (k0 + 64 < K) { loadA(k0 + 64); loadB(k0 + 64); }

    // ---- 2 WMMA K-substeps over the 64-wide LDS tile ----
#pragma unroll
    for (int ks = 0; ks < 64; ks += 32) {
      v16bf afr[2], bfr[2];
#pragma unroll
      for (int mi = 0; mi < 2; ++mi) {
        const int row = wm + mi*16 + lrow;
#pragma unroll
        for (int e = 0; e < 8; ++e) {
          afr[mi][e]   = sA[row][ks + kcol + e];
          afr[mi][e+8] = sA[row][ks + 16 + kcol + e];
        }
      }
#pragma unroll
      for (int ni = 0; ni < 2; ++ni) {
        const int col = wn + ni*16 + lrow;
#pragma unroll
        for (int e = 0; e < 8; ++e) {
          bfr[ni][e]   = sB[col][ks + kcol + e];
          bfr[ni][e+8] = sB[col][ks + 16 + kcol + e];
        }
      }
#pragma unroll
      for (int mi = 0; mi < 2; ++mi)
#pragma unroll
        for (int ni = 0; ni < 2; ++ni)
          acc[mi][ni] = __builtin_amdgcn_wmma_f32_16x16x32_bf16(
              false, afr[mi], false, bfr[ni], (short)0, acc[mi][ni], false, false);
    }
    __syncthreads();
  }

  // ---- epilogue: C/D layout: lanes0-15 M=e, lanes16-31 M=8+e ----
  const int rsel = (lane >> 4) * 8;
#pragma unroll
  for (int mi = 0; mi < 2; ++mi) {
#pragma unroll
    for (int ni = 0; ni < 2; ++ni) {
      const long col = tileN0 + wn + ni*16 + (lane & 15);
      const float bv = bias ? bias[col] : 0.0f;
#pragma unroll
      for (int e = 0; e < 8; ++e) {
        const long row = tileM0 + wm + mi*16 + rsel + e;
        C[row * (long)ldc + col] = acc[mi][ni][e] + bv;
      }
    }
  }
}

// ---------------------------------------------------------------------------
__global__ void embed_kernel(const int* __restrict__ ids,
                             const float* __restrict__ tok,
                             const float* __restrict__ pos,
                             float* __restrict__ x)
{
  const int bs = blockIdx.x;             // 0..M_-1
  const int s  = bs & (S_ - 1);
  const long id = ids[bs];
  for (int d = threadIdx.x; d < D_; d += blockDim.x)
    x[(long)bs*D_ + d] = tok[id*D_ + d] + pos[(long)s*D_ + d];
}

// y(row) = LN(x(row) + res(row)); in-place into x. One 256-thr block per row.
__global__ __launch_bounds__(256) void add_ln_kernel(
    float* __restrict__ x, const float* __restrict__ res,
    const float* __restrict__ g, const float* __restrict__ b)
{
  __shared__ float red[256];
  const int row = blockIdx.x, tid = threadIdx.x;
  float v[3]; float s = 0.f;
#pragma unroll
  for (int j = 0; j < 3; ++j) {
    const int d = tid + j*256;
    float t = x[(long)row*D_ + d];
    if (res) t += res[(long)row*D_ + d];
    v[j] = t; s += t;
  }
  red[tid] = s; __syncthreads();
  for (int o = 128; o > 0; o >>= 1) { if (tid < o) red[tid] += red[tid+o]; __syncthreads(); }
  const float mean = red[0] * (1.0f/D_);
  __syncthreads();
  float sq = 0.f;
#pragma unroll
  for (int j = 0; j < 3; ++j) { const float t = v[j]-mean; sq += t*t; }
  red[tid] = sq; __syncthreads();
  for (int o = 128; o > 0; o >>= 1) { if (tid < o) red[tid] += red[tid+o]; __syncthreads(); }
  const float rstd = rsqrtf(red[0]*(1.0f/D_) + 1e-5f);
#pragma unroll
  for (int j = 0; j < 3; ++j) {
    const int d = tid + j*256;
    x[(long)row*D_ + d] = (v[j]-mean)*rstd*g[d] + b[d];
  }
}

// scores[b,h,i,:] = softmax( (scores + R[i,(i-j)&511]) * SCALE, mask )
__global__ __launch_bounds__(256) void softmax_kernel(
    float* __restrict__ scores, const float* __restrict__ rbuf,
    const int* __restrict__ mask)
{
  __shared__ float red[256];
  const int i = blockIdx.x, h = blockIdx.y, b = blockIdx.z, tid = threadIdx.x;
  const long base = (((long)b*H_ + h)*S_ + i)*(long)S_;
  float v[2]; float mx = -3.4e38f;
#pragma unroll
  for (int j0 = 0; j0 < 2; ++j0) {
    const int j = tid + j0*256;
    float t = (scores[base + j] + rbuf[base + ((i - j) & (RELV-1))]) * 0.125f;
    if (mask[b*S_ + j] == 0) t = -1e9f;
    v[j0] = t; mx = fmaxf(mx, t);
  }
  red[tid] = mx; __syncthreads();
  for (int o = 128; o > 0; o >>= 1) { if (tid < o) red[tid] = fmaxf(red[tid], red[tid+o]); __syncthreads(); }
  mx = red[0]; __syncthreads();
  float sum = 0.f;
#pragma unroll
  for (int j0 = 0; j0 < 2; ++j0) { v[j0] = expf(v[j0] - mx); sum += v[j0]; }
  red[tid] = sum; __syncthreads();
  for (int o = 128; o > 0; o >>= 1) { if (tid < o) red[tid] += red[tid+o]; __syncthreads(); }
  const float inv = 1.0f / red[0];
#pragma unroll
  for (int j0 = 0; j0 < 2; ++j0) scores[base + tid + j0*256] = v[j0] * inv;
}

__global__ void gelu_kernel(float* __restrict__ h, long n)
{
  const long idx = (long)blockIdx.x * blockDim.x + threadIdx.x;
  if (idx < n) {
    const float x = h[idx];
    h[idx] = 0.5f * x * (1.0f + erff(x * 0.70710678118654752f));
  }
}

// ---------------------------------------------------------------------------
struct AttnP { const float *wqw,*wqb,*wkw,*wkb,*wvw,*wvb,*wow,*wob,*rel; };
struct EncP  { AttnP a; const float *g1,*b1,*g2,*b2,*f1w,*f1b,*f2w,*f2b; };
struct ReasP { AttnP a; const float *g,*b; };

static inline void launch_gemm(hipStream_t st, const float* A, const float* Bm,
    float* C, const float* bias, int Mrows, int N, int K,
    int lda, int ldb, int ldc,
    long aO, long aI, long bO, long bI, long cO, long cI,
    int inner, int batches, int bT)
{
  dim3 grid(N/64, Mrows/128, batches);
  gemm_bf16_wmma<<<grid, 256, 0, st>>>(A, Bm, C, bias, K, lda, ldb, ldc,
                                       aO, aI, bO, bI, cO, cI, inner, bT);
}

extern "C" void kernel_launch(void* const* d_in, const int* in_sizes, int n_in,
                              void* d_out, int out_size, void* d_ws, size_t ws_size,
                              hipStream_t stream)
{
  (void)in_sizes; (void)n_in; (void)out_size; (void)ws_size;

  const int* input_ids = (const int*)d_in[0];
  const int* amask     = (const int*)d_in[1];

  int p = 2;
  auto nextf = [&]() { return (const float*)d_in[p++]; };
  auto parse_attn = [&](AttnP& a) {
    a.wqw = nextf(); a.wqb = nextf();
    a.wkw = nextf(); a.wkb = nextf();
    a.wvw = nextf(); a.wvb = nextf();
    a.wow = nextf(); a.wob = nextf();
    a.rel = nextf();
  };

  const float* token_emb = nextf();
  const float* pos_emb   = nextf();
  EncP enc[NLAY];
  for (int i = 0; i < NLAY; ++i) {
    parse_attn(enc[i].a);
    enc[i].g1  = nextf(); enc[i].b1  = nextf();
    enc[i].g2  = nextf(); enc[i].b2  = nextf();
    enc[i].f1w = nextf(); enc[i].f1b = nextf();
    enc[i].f2w = nextf(); enc[i].f2b = nextf();
  }
  ReasP rea[NREAS];
  for (int i = 0; i < NREAS; ++i) {
    parse_attn(rea[i].a);
    rea[i].g = nextf(); rea[i].b = nextf();
  }
  const float* lnf_g = nextf();
  const float* lnf_b = nextf();
  const float* lm_w  = nextf();
  const float* lm_b  = nextf();

  // ---- workspace carve-up (floats) ----
  const long MD   = (long)M_ * D_;
  const long BHSS = (long)B_ * H_ * S_ * S_;
  float* w  = (float*)d_ws;
  float* x  = w; w += MD;     // residual stream
  float* t0 = w; w += MD;     // attn / ff output
  float* q  = w; w += MD;
  float* kb = w; w += MD;
  float* vb = w; w += MD;
  float* cx = w; w += MD;     // attention context
  float* sc = w; w += BHSS;   // scores / probs
  float* rb = w; w += BHSS;   // Q @ rel^T
  float* hb = w; w += (long)M_ * FF_;   // ff hidden

  embed_kernel<<<M_, 256, 0, stream>>>(input_ids, token_emb, pos_emb, x);

  const long SD = (long)S_ * D_;
  const long SS = (long)S_ * S_;

  auto run_mha = [&](const AttnP& a) {
    // Q,K,V projections: (2048x768)@(768x768)+b
    launch_gemm(stream, x, a.wqw, q,  a.wqb, M_, D_, D_, D_, D_, D_, 0,0,0,0,0,0, 1, 1, 0);
    launch_gemm(stream, x, a.wkw, kb, a.wkb, M_, D_, D_, D_, D_, D_, 0,0,0,0,0,0, 1, 1, 0);
    launch_gemm(stream, x, a.wvw, vb, a.wvb, M_, D_, D_, D_, D_, D_, 0,0,0,0,0,0, 1, 1, 0);
    // scores = Q @ K^T  per (b,h): A=q[b,:,h,:], B=k[b,:,h,:] (N-major => bT=1)
    launch_gemm(stream, q, kb, sc, nullptr, S_, S_, DK_, D_, D_, S_,
                SD, DK_, SD, DK_, (long)H_*SS, SS, H_, B_*H_, 1);
    // R = Q @ rel^T  per (b,h): B=rel (512x64, shared across batches)
    launch_gemm(stream, q, a.rel, rb, nullptr, S_, RELV, DK_, D_, DK_, S_,
                SD, DK_, 0, 0, (long)H_*SS, SS, H_, B_*H_, 1);
    softmax_kernel<<<dim3(S_, H_, B_), 256, 0, stream>>>(sc, rb, amask);
    // ctx = P @ V  per (b,h): A=probs (512x512), B=v[b,:,h,:] (K-major => bT=0)
    launch_gemm(stream, sc, vb, cx, nullptr, S_, DK_, S_, S_, D_, D_,
                (long)H_*SS, SS, SD, DK_, SD, DK_, H_, B_*H_, 0);
    // out proj
    launch_gemm(stream, cx, a.wow, t0, a.wob, M_, D_, D_, D_, D_, D_, 0,0,0,0,0,0, 1, 1, 0);
  };

  for (int L = 0; L < NLAY; ++L) {
    run_mha(enc[L].a);
    add_ln_kernel<<<M_, 256, 0, stream>>>(x, t0, enc[L].g1, enc[L].b1);
    launch_gemm(stream, x, enc[L].f1w, hb, enc[L].f1b, M_, FF_, D_, D_, FF_, FF_,
                0,0,0,0,0,0, 1, 1, 0);
    const long n = (long)M_ * FF_;
    gelu_kernel<<<(int)((n + 255) / 256), 256, 0, stream>>>(hb, n);
    launch_gemm(stream, hb, enc[L].f2w, t0, enc[L].f2b, M_, D_, FF_, FF_, D_, D_,
                0,0,0,0,0,0, 1, 1, 0);
    add_ln_kernel<<<M_, 256, 0, stream>>>(x, t0, enc[L].g2, enc[L].b2);
  }

  for (int L = 0; L < NREAS; ++L) {
    run_mha(rea[L].a);
    add_ln_kernel<<<M_, 256, 0, stream>>>(x, t0, rea[L].g, rea[L].b);
  }

  add_ln_kernel<<<M_, 256, 0, stream>>>(x, nullptr, lnf_g, lnf_b);

  // logits = x @ lm_w + lm_b : (2048x768)@(768x32000)
  launch_gemm(stream, x, lm_w, (float*)d_out, lm_b, M_, V_, D_, D_, V_, V_,
              0,0,0,0,0,0, 1, 1, 0);
}
